// EvoAttentionCausalTorch_13340168421957
// MI455X (gfx1250) — matrix operand order repaired and checked by reading, not compiled, not run on hardware
//
#include <hip/hip_runtime.h>

#define DD    128
#define LL    4096
#define CHUNK 64
#define BLOCK 256
#define FSTR  132   // fp32 chunk-buffer row stride (floats), padded (matches TDM pad)
#define WSTR  136   // transposed-weight bf16 row stride, padded (16B-aligned rows)
#define ASTR  136   // activation bf16 row stride, padded (16B-aligned rows)

typedef __attribute__((ext_vector_type(16))) __bf16 bf16x16;
typedef __attribute__((ext_vector_type(8)))  __bf16 bf16x8;
typedef __attribute__((ext_vector_type(8)))  float  f32x8;
typedef __attribute__((ext_vector_type(4)))  unsigned u32x4;
typedef __attribute__((ext_vector_type(8)))  int      i32x8;

union ABfrag { bf16x16 v; bf16x8 h[2]; };

__device__ __forceinline__ unsigned bfbits(float f) {
  union { float f; unsigned u; } v; v.f = f;
  return (v.u + 0x7FFFu + ((v.u >> 16) & 1u)) >> 16;   // RNE
}
__device__ __forceinline__ __bf16 f2bf(float f) {
  unsigned short h = (unsigned short)bfbits(f);
  union { unsigned short s; __bf16 b; } o; o.s = h; return o.b;
}
__device__ __forceinline__ unsigned pk2(float a, float b) {
  return bfbits(a) | (bfbits(b) << 16);
}
__device__ __forceinline__ float rcpf(float x) { return __builtin_amdgcn_rcpf(x); }
__device__ __forceinline__ float sigm(float x) { return rcpf(1.0f + __expf(-x)); }

// TDM: async DMA of a 64x128 fp32 tile (row stride 128) from global into LDS,
// hardware-padded to 132-DWORD rows (4 pad DWORDs after every 128 DWORDs).
__device__ __forceinline__ void tdm_load_chunk(const float* gsrc, unsigned lds_off) {
  unsigned long long ga = (unsigned long long)(uintptr_t)gsrc;
  u32x4 g0;
  g0[0] = 1u;                                            // count=1 (valid), user mode
  g0[1] = lds_off;                                       // lds_addr (bytes)
  g0[2] = (unsigned)(ga & 0xFFFFFFFFull);                // global_addr[31:0]
  g0[3] = (unsigned)((ga >> 32) & 0x01FFFFFFull)         // global_addr[56:32]
        | (2u << 30);                                    // type=2 ("image")
  i32x8 g1;
  g1[0] = (int)((2u << 16)      // data_size = 4 bytes
              | (1u << 20)      // pad_enable
              | (6u << 22)      // pad_interval: 2^(6+1)=128 DWORDs
              | (3u << 25));    // pad_amount: 3+1 = 4 DWORDs
  g1[1] = (int)(128u << 16);    // tensor_dim0 = 128 (bits 79:48)
  g1[2] = (int)(64u  << 16);    // tensor_dim1 = 64  (bits 111:80)
  g1[3] = (int)(128u << 16);    // tile_dim0   = 128 (bits 127:112)
  g1[4] = 64;                   // tile_dim1   = 64  (bits 143:128)
  g1[5] = 128;                  // tensor_dim0_stride = 128 (bits 207:160)
  g1[6] = 0;
  g1[7] = 0;
  asm volatile("tensor_load_to_lds %0, %1" :: "s"(g0), "s"(g1) : "memory");
}

__global__ __launch_bounds__(BLOCK) void evo_attn_kernel(
    const float* __restrict__ V,
    const float* __restrict__ Wq, const float* __restrict__ bq,
    const float* __restrict__ Wv, const float* __restrict__ bv,
    const float* __restrict__ Wg, const float* __restrict__ bg,
    const float* __restrict__ Wf, const float* __restrict__ bf_,
    const float* __restrict__ gamma, const float* __restrict__ beta,
    float* __restrict__ out)
{
  // ---- LDS ----
  __shared__ __bf16 sWqT[DD * WSTR];   // transposed weights: [n][k] bf16
  __shared__ __bf16 sWvT[DD * WSTR];
  __shared__ __bf16 sWgT[DD * WSTR];
  __shared__ __bf16 sWfT[DD * WSTR];
  __shared__ float  sVf [CHUNK * FSTR];   // V chunk fp32 (TDM destination)
  __shared__ float  sCtx[CHUNK * FSTR];   // context fp32
  __shared__ float  sStat[CHUNK * FSTR];  // stat, later out_pre
  __shared__ __bf16 sVb[CHUNK * ASTR];    // V chunk bf16 (A of GEMM2/3)
  __shared__ __bf16 sAb[CHUNK * ASTR];    // mean bf16 (A of GEMM1) -> gate bf16 (A of GEMM4)
  __shared__ float  sbq[DD], sbv[DD], sbg[DD], sbf[DD], sgam[DD], sbet[DD];
  __shared__ float  cctx[DD], cstat[DD];
  __shared__ float  sAlive[CHUNK], sInv[CHUNK];

  const int tid  = threadIdx.x;
  const int lane = tid & 31;
  const int wave = tid >> 5;           // 8 waves -> 8 N-tiles
  const int half = lane >> 4;
  const int l16  = lane & 15;
  const int ncol = wave * 16 + l16;
  const size_t baseoff = (size_t)blockIdx.x * LL * DD;
  const unsigned sVf_lds = (unsigned)(uintptr_t)(void*)&sVf[0];  // LDS byte offset

  // ---- preload weights transposed (fp32 [k][n] -> bf16 [n][k]) + vectors ----
  for (int i = tid; i < DD * DD; i += BLOCK) {
    int k = i >> 7, n = i & 127;
    sWqT[n * WSTR + k] = f2bf(Wq[i]);
    sWvT[n * WSTR + k] = f2bf(Wv[i]);
    sWgT[n * WSTR + k] = f2bf(Wg[i]);
    sWfT[n * WSTR + k] = f2bf(Wf[i]);
  }
  if (tid < DD) {
    sbq[tid] = bq[tid]; sbv[tid] = bv[tid]; sbg[tid] = bg[tid]; sbf[tid] = bf_[tid];
    sgam[tid] = gamma[tid]; sbet[tid] = beta[tid];
    cctx[tid] = 0.0f; cstat[tid] = 0.0f;
  }
  // kick off the async DMA of the first V chunk
  if (wave == 0) tdm_load_chunk(V + baseoff, sVf_lds);
  __syncthreads();

  for (int cb = 0; cb < LL; cb += CHUNK) {
    // ---- wait for TDM tile, then make bf16 copy + 1/(idx+1) table ----
    if (wave == 0) __builtin_amdgcn_s_wait_tensorcnt(0);
    __syncthreads();
    for (int i = tid; i < CHUNK * DD / 4; i += BLOCK) {
      int t = i >> 5, d4 = i & 31;
      float4 v = *(const float4*)&sVf[t * FSTR + d4 * 4];
      uint2 w; w.x = pk2(v.x, v.y); w.y = pk2(v.z, v.w);
      *(uint2*)&sVb[t * ASTR + d4 * 4] = w;
    }
    if (tid < CHUNK) sInv[tid] = rcpf((float)(cb + tid + 1));
    __syncthreads();

    // ---- context cumsum -> sCtx(fp32) + mean bf16 -> sAb; alive flags ----
    if (tid < DD) {
      float carry = cctx[tid];
      for (int t = 0; t < CHUNK; ++t) {
        carry += sVf[t * FSTR + tid];
        sCtx[t * FSTR + tid] = carry;
        sAb[t * ASTR + tid] = f2bf(carry * sInv[t]);
      }
      cctx[tid] = carry;
    } else if (tid < DD + CHUNK) {
      int t = tid - DD;
      float s = 0.0f;
      for (int d = 0; d < DD; ++d) s += fabsf(sVf[t * FSTR + d]);
      sAlive[t] = (s > 0.0f) ? 1.0f : 0.0f;
    }
    __syncthreads();

    // ---- GEMM1: G = sigmoid(mean @ Wg + bg); sStat = G*V ----
    {
      ABfrag bw[4];
#pragma unroll
      for (int kt = 0; kt < 4; ++kt) {
        const __bf16* p = sWgT + ncol * WSTR + kt * 32 + 16 * half;
        bw[kt].h[0] = *(const bf16x8*)p;
        bw[kt].h[1] = *(const bf16x8*)(p + 8);
      }
      for (int mt = 0; mt < 4; ++mt) {
        f32x8 acc = {0.f,0.f,0.f,0.f,0.f,0.f,0.f,0.f};
#pragma unroll
        for (int kt = 0; kt < 4; ++kt) {
          ABfrag a;
          const __bf16* p = sAb + (mt * 16 + l16) * ASTR + kt * 32 + 16 * half;
          a.h[0] = *(const bf16x8*)p;
          a.h[1] = *(const bf16x8*)(p + 8);
          acc = __builtin_amdgcn_wmma_f32_16x16x32_bf16(false, a.v, false, bw[kt].v,
                                                        (short)0, acc, false, false);
        }
#pragma unroll
        for (int r = 0; r < 8; ++r) {
          int t = mt * 16 + 8 * half + r;
          float g = sigm(acc[r] + sbg[ncol]);
          sStat[t * FSTR + ncol] = g * sVf[t * FSTR + ncol];
        }
      }
    }
    __syncthreads();

    // ---- stat cumsum (carry across chunks) ----
    if (tid < DD) {
      float carry = cstat[tid];
      for (int t = 0; t < CHUNK; ++t) {
        carry += sStat[t * FSTR + tid];
        sStat[t * FSTR + tid] = carry;
      }
      cstat[tid] = carry;
    }
    __syncthreads();

    // ---- GEMM2+3 fused (shared A = V bf16): local = V@Wv, q = V@Wq ----
    {
      ABfrag bwv[4], bwq[4];
#pragma unroll
      for (int kt = 0; kt < 4; ++kt) {
        const __bf16* pv = sWvT + ncol * WSTR + kt * 32 + 16 * half;
        const __bf16* pq = sWqT + ncol * WSTR + kt * 32 + 16 * half;
        bwv[kt].h[0] = *(const bf16x8*)pv;  bwv[kt].h[1] = *(const bf16x8*)(pv + 8);
        bwq[kt].h[0] = *(const bf16x8*)pq;  bwq[kt].h[1] = *(const bf16x8*)(pq + 8);
      }
      for (int mt = 0; mt < 4; ++mt) {
        f32x8 accL = {0.f,0.f,0.f,0.f,0.f,0.f,0.f,0.f};
        f32x8 accQ = {0.f,0.f,0.f,0.f,0.f,0.f,0.f,0.f};
#pragma unroll
        for (int kt = 0; kt < 4; ++kt) {
          ABfrag a;
          const __bf16* p = sVb + (mt * 16 + l16) * ASTR + kt * 32 + 16 * half;
          a.h[0] = *(const bf16x8*)p;
          a.h[1] = *(const bf16x8*)(p + 8);
          accL = __builtin_amdgcn_wmma_f32_16x16x32_bf16(false, a.v, false, bwv[kt].v,
                                                         (short)0, accL, false, false);
          accQ = __builtin_amdgcn_wmma_f32_16x16x32_bf16(false, a.v, false, bwq[kt].v,
                                                         (short)0, accQ, false, false);
        }
        // epilogue: gate = silu(q/den) * V -> sAb (bf16)
#pragma unroll
        for (int r = 0; r < 8; ++r) {
          int t = mt * 16 + 8 * half + r;
          float localv = accL[r] + sbv[ncol];
          float qv     = accQ[r] + sbq[ncol];
          float den = fabsf(sStat[t * FSTR + ncol]) + fabsf(localv) + 1e-8f;
          float qn  = qv * rcpf(den);
          float gate = (qn * sigm(qn)) * sVf[t * FSTR + ncol];
          sAb[t * ASTR + ncol] = f2bf(gate);
        }
      }
    }
    __syncthreads();

    // ---- sVf is dead from here: start async DMA of the next V chunk ----
    if (wave == 0 && cb + CHUNK < LL)
      tdm_load_chunk(V + baseoff + (size_t)(cb + CHUNK) * DD, sVf_lds);

    // ---- GEMM4: f = sigmoid(gate @ Wf + bf); out_pre = f*ctx*alive -> sStat ----
    {
      ABfrag bw[4];
#pragma unroll
      for (int kt = 0; kt < 4; ++kt) {
        const __bf16* p = sWfT + ncol * WSTR + kt * 32 + 16 * half;
        bw[kt].h[0] = *(const bf16x8*)p;
        bw[kt].h[1] = *(const bf16x8*)(p + 8);
      }
      for (int mt = 0; mt < 4; ++mt) {
        f32x8 acc = {0.f,0.f,0.f,0.f,0.f,0.f,0.f,0.f};
#pragma unroll
        for (int kt = 0; kt < 4; ++kt) {
          ABfrag a;
          const __bf16* p = sAb + (mt * 16 + l16) * ASTR + kt * 32 + 16 * half;
          a.h[0] = *(const bf16x8*)p;
          a.h[1] = *(const bf16x8*)(p + 8);
          acc = __builtin_amdgcn_wmma_f32_16x16x32_bf16(false, a.v, false, bw[kt].v,
                                                        (short)0, acc, false, false);
        }
#pragma unroll
        for (int r = 0; r < 8; ++r) {
          int t = mt * 16 + 8 * half + r;
          float f = sigm(acc[r] + sbf[ncol]);
          sStat[t * FSTR + ncol] = f * sCtx[t * FSTR + ncol] * sAlive[t];
        }
      }
    }
    __syncthreads();

    // ---- LayerNorm over d=128 (4 lanes/row, shuffle reduce) + store ----
    {
      int row = tid >> 2, sub = tid & 3;
      float s = 0.0f, ss = 0.0f;
#pragma unroll
      for (int j4 = 0; j4 < 8; ++j4) {
        float4 x = *(const float4*)&sStat[row * FSTR + sub * 32 + j4 * 4];
        s  += x.x + x.y + x.z + x.w;
        ss += x.x*x.x + x.y*x.y + x.z*x.z + x.w*x.w;
      }
      s  += __shfl_xor(s, 1, 32);  ss += __shfl_xor(ss, 1, 32);
      s  += __shfl_xor(s, 2, 32);  ss += __shfl_xor(ss, 2, 32);
      float mu  = s * (1.0f / 128.0f);
      float var = ss * (1.0f / 128.0f) - mu * mu;
      float rs  = __builtin_amdgcn_rsqf(var + 1e-5f);
      size_t ob = baseoff + (size_t)(cb + row) * DD + sub * 32;
#pragma unroll
      for (int j4 = 0; j4 < 8; ++j4) {
        int c = sub * 32 + j4 * 4;
        float4 x = *(const float4*)&sStat[row * FSTR + c];
        float4 g = *(const float4*)&sgam[c];
        float4 b = *(const float4*)&sbet[c];
        float4 o;
        o.x = (x.x - mu) * rs * g.x + b.x;
        o.y = (x.y - mu) * rs * g.y + b.y;
        o.z = (x.z - mu) * rs * g.z + b.z;
        o.w = (x.w - mu) * rs * g.w + b.w;
        *(float4*)(out + ob + j4 * 4) = o;
      }
    }
    __syncthreads();
  }
}

extern "C" void kernel_launch(void* const* d_in, const int* in_sizes, int n_in,
                              void* d_out, int out_size, void* d_ws, size_t ws_size,
                              hipStream_t stream) {
  const float* V     = (const float*)d_in[0];
  const float* Wq    = (const float*)d_in[1];
  const float* bq    = (const float*)d_in[2];
  const float* Wv    = (const float*)d_in[3];
  const float* bv    = (const float*)d_in[4];
  const float* Wg    = (const float*)d_in[5];
  const float* bg    = (const float*)d_in[6];
  const float* Wf    = (const float*)d_in[7];
  const float* bf_   = (const float*)d_in[8];
  const float* gamma = (const float*)d_in[9];
  const float* beta  = (const float*)d_in[10];
  float* out = (float*)d_out;

  const int bh = in_sizes[0] / (LL * DD);   // b*h independent sequences
  hipLaunchKernelGGL(evo_attn_kernel, dim3(bh), dim3(BLOCK), 0, stream,
                     V, Wq, bq, Wv, bv, Wg, bg, Wf, bf_, gamma, beta, out);
}
